// Transformer_1211180777724
// MI455X (gfx1250) — matrix operand order repaired and checked by reading, not compiled, not run on hardware
//
#include <hip/hip_runtime.h>
#include <hip/hip_bf16.h>

// ---------------------------------------------------------------------------
// Transformer block (pre-LN attention + FFN) for MI455X (gfx1250, wave32).
// GEMMs + attention use v_wmma_f32_16x16x32_f16. Tile staging uses the CDNA5
// data-movement paths: TDM (TENSOR_LOAD_TO_LDS + s_wait_tensorcnt) for the
// GEMM A tiles, async global->LDS (GLOBAL_LOAD_ASYNC_TO_LDS_B128 +
// s_wait_asynccnt) for the attention K tiles.
// ---------------------------------------------------------------------------

typedef __attribute__((ext_vector_type(16))) _Float16 v16h;
typedef __attribute__((ext_vector_type(8)))  _Float16 v8h;
typedef __attribute__((ext_vector_type(8)))  float    v8f;
typedef __attribute__((ext_vector_type(4)))  unsigned int v4u;
typedef __attribute__((ext_vector_type(8)))  int      v8i;
typedef __attribute__((ext_vector_type(4)))  int      v4i;

#define Bc   2
#define Tc   2048
#define Ec   1024
#define Hc   16
#define DVc  64
#define E4c  4096
#define BTc  4096              // B*T rows
#define ATT_SCALE (1.0f/32.0f) // E^-0.5 (reference scales by full embed dim)

// ---- CDNA5 async global->LDS copy (ASYNCcnt-tracked) -----------------------

static __device__ __forceinline__ void async_copy_b128(void* lds_dst,
                                                       const void* gsrc) {
  unsigned lds = (unsigned)(size_t)lds_dst;          // flat LDS addr low bits
  unsigned long long g = (unsigned long long)(size_t)gsrc;
  asm volatile("global_load_async_to_lds_b128 %0, %1, off"
               :: "v"(lds), "v"(g) : "memory");
}
static __device__ __forceinline__ void wait_async0() {
  asm volatile("s_wait_asynccnt 0x0" ::: "memory");
}
static __device__ __forceinline__ void wait_tensor0() {
#if __has_builtin(__builtin_amdgcn_s_wait_tensorcnt)
  __builtin_amdgcn_s_wait_tensorcnt(0);
#else
  asm volatile("s_wait_tensorcnt 0x0" ::: "memory");
#endif
}

// TDM: DMA a rows x cols (f16) tile from a row-major tensor (row stride
// `stride` halves) at gsrc into contiguous LDS at lds_dst. 2D descriptor per
// CDNA5 ISA; issued by the calling wave only.
// This toolchain exposes the 6-arg builtin form:
//   (uint32x4 g0, int32x8 g1, int32x4 g2, int32x4 g3, int32x8, i32 cpol)
static __device__ __forceinline__ void tdm_load_tile_f16(
    void* lds_dst, const void* gsrc, int rows, int cols, int stride,
    int tdim0, int tdim1) {
#if __has_builtin(__builtin_amdgcn_tensor_load_to_lds)
  unsigned lds = (unsigned)(size_t)lds_dst;
  unsigned long long ga = (unsigned long long)(size_t)gsrc;
  v4u g0;
  g0.x = 1u;                                        // count=1, user mode
  g0.y = lds;                                       // lds_addr (bytes)
  g0.z = (unsigned)(ga & 0xffffffffu);              // global_addr[31:0]
  g0.w = (unsigned)((ga >> 32) & 0x1ffffffu)        // global_addr[56:32]
         | (2u << 30);                              // type=2 ("image")
  v8i g1;
  g1[0] = (int)(1u << 16);                          // data_size=1 -> 2 bytes
  g1[1] = (int)(((unsigned)tdim0 & 0xffffu) << 16); // tensor_dim0[15:0]
  g1[2] = (int)((((unsigned)tdim0 >> 16) & 0xffffu) |
                (((unsigned)tdim1 & 0xffffu) << 16));     // dim0 hi, dim1 lo
  g1[3] = (int)((((unsigned)tdim1 >> 16) & 0xffffu) |
                (((unsigned)cols & 0xffffu) << 16));      // dim1 hi, tile_dim0
  g1[4] = (int)((unsigned)rows & 0xffffu);          // tile_dim1 (tile_dim2=0)
  g1[5] = (int)stride;                              // tensor_dim0_stride lo32
  g1[6] = 0;                                        // stride hi, dim1_stride lo
  g1[7] = 0;
  v4i gz4 = {0, 0, 0, 0};
  v8i gz8 = {0, 0, 0, 0, 0, 0, 0, 0};
  __builtin_amdgcn_tensor_load_to_lds(g0, g1, gz4, gz4, gz8, 0);
#else
  // Fallback: per-lane async b128 copies.
  int lane = threadIdx.x & 31;
  for (int r = lane; r < rows; r += 32)
    for (int c = 0; c < cols; c += 8)
      async_copy_b128((_Float16*)lds_dst + r * cols + c,
                      (const _Float16*)gsrc + (size_t)r * stride + c);
  wait_async0();
#endif
}

// ---- WMMA helpers ----------------------------------------------------------

static __device__ __forceinline__ v8f wmma_f16(v16h a, v16h b, v8f c) {
  // (neg_a, A, neg_b, B, c_mod, C, reuse_a, reuse_b)
  return __builtin_amdgcn_wmma_f32_16x16x32_f16(false, a, false, b, (short)0, c,
                                                false, false);
}

// A-operand 16x32 f16 fragment: lane holds row (lane&15); elements 0..15
// are k = hi*16 .. hi*16+15 (contiguous, hi = lane>>4).
static __device__ __forceinline__ v16h load_frag_a(const _Float16* p0,
                                                   const _Float16* p1) {
  v8h a = *(const v8h*)p0;
  v8h b = *(const v8h*)p1;
  v16h r;
#pragma unroll
  for (int i = 0; i < 8; ++i) { r[i] = a[i]; r[i + 8] = b[i]; }
  return r;
}

// B-operand 32x16 f16 fragment: lane holds column (lane&15); elements 0..15
// are k = hi*16 .. hi*16+15 (contiguous).
static __device__ __forceinline__ v16h load_frag_b(const _Float16* p) {
  v8h a = *(const v8h*)p;
  v8h b = *(const v8h*)(p + 8);
  v16h r;
#pragma unroll
  for (int i = 0; i < 8; ++i) { r[i] = a[i]; r[i + 8] = b[i]; }
  return r;
}

// ---- weight repack kernels -------------------------------------------------

// wq/wk/wv: (H, E, DV) f32  ->  wqkv16: [3][E][E] f16 with col = h*DV + d
__global__ __launch_bounds__(256) void repack_qkv_kernel(
    const float* __restrict__ wq, const float* __restrict__ wk,
    const float* __restrict__ wv, _Float16* __restrict__ out) {
  int idx = blockIdx.x * 256 + threadIdx.x;           // < 3*E*E
  int w   = idx / (Ec * Ec);
  int rem = idx - w * (Ec * Ec);
  int e   = rem >> 10;                                 // / E
  int col = rem & (Ec - 1);
  int h   = col >> 6;                                  // / DV
  int d   = col & (DVc - 1);
  const float* src = (w == 0) ? wq : (w == 1) ? wk : wv;
  out[idx] = (_Float16)src[((size_t)h * Ec + e) * DVc + d];
}

__global__ __launch_bounds__(256) void cvt_f32_f16_kernel(
    const float* __restrict__ in, _Float16* __restrict__ out, int n) {
  int i = blockIdx.x * 256 + threadIdx.x;
  if (i < n) out[i] = (_Float16)in[i];
}

// ---- LayerNorm: one 256-thread block per row of E=1024 --------------------

__global__ __launch_bounds__(256) void ln_kernel(
    const float* __restrict__ x, const float* __restrict__ g,
    const float* __restrict__ beta, _Float16* __restrict__ out) {
  __shared__ float s_sum[8], s_sq[8];
  const int row = blockIdx.x;
  const float* xr = x + (size_t)row * Ec;
  float vals[4];
  float sum = 0.f, sq = 0.f;
#pragma unroll
  for (int i = 0; i < 4; ++i) {
    float v = xr[threadIdx.x + i * 256];
    vals[i] = v; sum += v; sq += v * v;
  }
#pragma unroll
  for (int m = 1; m < 32; m <<= 1) {
    sum += __shfl_xor(sum, m, 32);
    sq  += __shfl_xor(sq, m, 32);
  }
  int wid = threadIdx.x >> 5;
  if ((threadIdx.x & 31) == 0) { s_sum[wid] = sum; s_sq[wid] = sq; }
  __syncthreads();
  float tsum = 0.f, tsq = 0.f;
#pragma unroll
  for (int w = 0; w < 8; ++w) { tsum += s_sum[w]; tsq += s_sq[w]; }
  const float inv = 1.0f / (float)Ec;
  float mean = tsum * inv;
  float var  = tsq * inv - mean * mean;
  float rstd = rsqrtf(var + 1e-5f);
#pragma unroll
  for (int i = 0; i < 4; ++i) {
    int c = threadIdx.x + i * 256;
    out[(size_t)row * Ec + c] = (_Float16)((vals[i] - mean) * rstd * g[c] + beta[c]);
  }
}

// ---- WMMA GEMM: C[M,N] = A[M,K] * B[K,N]  (f16 in, f32 acc) ---------------
// EPI 0: store f16 (no bias)          -> QKV projections
// EPI 1: +bias, ReLU, store f16       -> FFN1
// EPI 2: +bias, +residual, store f32  -> out-proj, FFN2
#define BM 64
#define BN 64
#define BK 64   // two WMMA k-steps per LDS stage

template <int EPI>
__global__ __launch_bounds__(128) void gemm_kernel(
    const _Float16* __restrict__ A, const _Float16* __restrict__ Bm,
    void* __restrict__ Cout, const float* __restrict__ bias,
    const float* __restrict__ resid, int M, int N, int K) {
  __shared__ __align__(16) _Float16 Ash[BM][BK];       // 8 KB (TDM dest)
  __shared__ __align__(16) _Float16 Bts[BN][BK + 8];   // 9 KB, transposed [n][k]

  const int tid  = threadIdx.x;
  const int lane = tid & 31;
  const int wave = tid >> 5;                // 4 waves
  const int tileM = blockIdx.y * BM;
  const int tileN = blockIdx.x * BN;
  const int wrow = (wave >> 1) * 32;        // wave subtile 32x32
  const int wcol = (wave & 1) * 32;
  const int mloc = lane & 15;
  const int hi   = lane >> 4;

  v8f zero = {};
  v8f acc[2][2];
  acc[0][0] = zero; acc[0][1] = zero; acc[1][0] = zero; acc[1][1] = zero;

  const int bk = tid >> 1;                  // B stage k-row 0..63
  const int bn = (tid & 1) * 32;            // B stage n 0 / 32

  for (int k0 = 0; k0 < K; k0 += BK) {
    __syncthreads();
    // --- stage A tile (64x64) via the Tensor Data Mover, wave 0 only ---
    if (wave == 0) {
      tdm_load_tile_f16(&Ash[0][0], A + (size_t)tileM * K + k0,
                        /*rows=*/BM, /*cols=*/BK, /*stride=*/K,
                        /*tdim0=*/K, /*tdim1=*/M);
    }
    // --- stage B tile (64x64) transposed into Bts[n][k] (VGPR roundtrip) ---
    {
      const _Float16* src = Bm + (size_t)(k0 + bk) * N + tileN + bn;
      v8h bv[4];
#pragma unroll
      for (int c = 0; c < 4; ++c) bv[c] = *(const v8h*)(src + c * 8);
#pragma unroll
      for (int c = 0; c < 4; ++c)
#pragma unroll
        for (int j = 0; j < 8; ++j) Bts[bn + c * 8 + j][bk] = bv[c][j];
    }
    if (k0 + BK < K)
      __builtin_prefetch(Bm + (size_t)(k0 + BK + bk) * N + tileN + bn, 0, 1);
    if (wave == 0) wait_tensor0();          // A tile DMA complete
    __syncthreads();                        // publish A (TDM) + B (ds) tiles

#pragma unroll
    for (int kk = 0; kk < BK; kk += 32) {
      v16h afr[2], bfr[2];
#pragma unroll
      for (int fi = 0; fi < 2; ++fi) {
        int r = wrow + 16 * fi + mloc;
        afr[fi] = load_frag_a(&Ash[r][kk + hi * 16], &Ash[r][kk + hi * 16 + 8]);
      }
#pragma unroll
      for (int fj = 0; fj < 2; ++fj) {
        int n = wcol + 16 * fj + mloc;
        bfr[fj] = load_frag_b(&Bts[n][kk + hi * 16]);
      }
#pragma unroll
      for (int i = 0; i < 2; ++i)
#pragma unroll
        for (int j = 0; j < 2; ++j)
          acc[i][j] = wmma_f16(afr[i], bfr[j], acc[i][j]);
    }
  }

  // --- epilogue ---
#pragma unroll
  for (int i = 0; i < 2; ++i) {
#pragma unroll
    for (int j = 0; j < 2; ++j) {
#pragma unroll
      for (int r = 0; r < 8; ++r) {
        int gm = tileM + wrow + 16 * i + r + 8 * hi;
        int gn = tileN + wcol + 16 * j + mloc;
        float val = acc[i][j][r];
        size_t o = (size_t)gm * N + gn;
        if (EPI == 0) {
          ((_Float16*)Cout)[o] = (_Float16)val;
        } else if (EPI == 1) {
          val += bias[gn];
          val = fmaxf(val, 0.0f);
          ((_Float16*)Cout)[o] = (_Float16)val;
        } else {
          val += bias[gn] + resid[o];
          ((float*)Cout)[o] = val;
        }
      }
    }
  }
}

// ---- Flash attention: 4 waves / block, 64 query rows, shared K/V tiles ----
// q/k/v/o layout: [B, T, H, DV] f16 (row t contiguous over DV per head).

__global__ __launch_bounds__(128) void attn_kernel(
    const _Float16* __restrict__ q, const _Float16* __restrict__ k,
    const _Float16* __restrict__ v, _Float16* __restrict__ o) {
  __shared__ __align__(16) _Float16 Ksh[32][72];      // K tile [s][d], 4.5 KB
  __shared__ __align__(16) _Float16 Vt[DVc][40];      // V tile transposed [d][s]
  __shared__ __align__(16) _Float16 Pt[4][16][32];    // per-wave probs [m][s]

  const int tid  = threadIdx.x;
  const int lane = tid & 31;
  const int wave = tid >> 5;
  const int mloc = lane & 15;
  const int hi   = lane >> 4;
  const int bid  = blockIdx.x;
  const int tb   = (bid & (Tc / 64 - 1)) * 64;        // block query-tile base
  const int bh   = bid >> 5;                          // T/64 = 32
  const int b    = bh >> 4;
  const int h    = bh & (Hc - 1);
  const int tq0  = tb + wave * 16;                    // this wave's query rows

  auto rowptr = [&](const _Float16* base, int t) {
    return base + ((size_t)((b * Tc + t) * Hc + h)) * DVc;
  };

  // Q fragments (A layout), d-chunks 0..31 and 32..63.
  const _Float16* qr = rowptr(q, tq0 + mloc);
  v16h aq0 = load_frag_a(qr + hi * 16, qr + hi * 16 + 8);
  v16h aq1 = load_frag_a(qr + 32 + hi * 16, qr + 40 + hi * 16);

  float mrun[8], lrun[8];
  v8f zero = {};
  v8f oacc[4];
#pragma unroll
  for (int r = 0; r < 8; ++r) { mrun[r] = -1e30f; lrun[r] = 0.0f; }
#pragma unroll
  for (int dt = 0; dt < 4; ++dt) oacc[dt] = zero;

  const int kr = tid >> 2;            // K/V stage row 0..31
  const int kc = (tid & 3) * 16;      // stage col group

  const int send = tb + 64;           // causal bound for the whole block
  for (int s0 = 0; s0 < send; s0 += 32) {
    __syncthreads();                  // protect LDS reuse across iterations
    // --- stage K tile via async global->LDS (row-major, 2x b128/thread) ---
    {
      const _Float16* src = rowptr(k, s0 + kr) + kc;
      async_copy_b128(&Ksh[kr][kc], src);
      async_copy_b128(&Ksh[kr][kc + 8], src + 8);
    }
    // --- stage V tile transposed (VGPR roundtrip) ---
    {
      const _Float16* src = rowptr(v, s0 + kr) + kc;
      v8h c0 = *(const v8h*)src;
      v8h c1 = *(const v8h*)(src + 8);
#pragma unroll
      for (int j = 0; j < 8; ++j) {
        Vt[kc + j][kr]     = c0[j];
        Vt[kc + 8 + j][kr] = c1[j];
      }
    }
    wait_async0();
    __syncthreads();

    // --- scores: S = Q * K^T, two 16-col fragments (B operand from LDS) ---
    v16h bk00 = load_frag_b(&Ksh[mloc][hi * 16]);
    v16h bk01 = load_frag_b(&Ksh[mloc][32 + hi * 16]);
    v16h bk10 = load_frag_b(&Ksh[16 + mloc][hi * 16]);
    v16h bk11 = load_frag_b(&Ksh[16 + mloc][32 + hi * 16]);
    v8f S0 = wmma_f16(aq0, bk00, zero);
    S0     = wmma_f16(aq1, bk01, S0);
    v8f S1 = wmma_f16(aq0, bk10, zero);
    S1     = wmma_f16(aq1, bk11, S1);

    // --- online softmax update (f32), C-frag row m = r + 8*hi ---
    float alpha[8];
#pragma unroll
    for (int r = 0; r < 8; ++r) {
      int mg  = tq0 + r + 8 * hi;
      int sg0 = s0 + mloc;
      int sg1 = s0 + 16 + mloc;
      float a0 = (sg0 <= mg) ? S0[r] * ATT_SCALE : -1e30f;
      float a1 = (sg1 <= mg) ? S1[r] * ATT_SCALE : -1e30f;
      float rm = fmaxf(a0, a1);
#pragma unroll
      for (int m = 1; m < 16; m <<= 1) rm = fmaxf(rm, __shfl_xor(rm, m, 32));
      float mnew = fmaxf(mrun[r], rm);
      float al   = __expf(mrun[r] - mnew);
      float p0   = __expf(a0 - mnew);
      float p1   = __expf(a1 - mnew);
      float rs   = p0 + p1;
#pragma unroll
      for (int m = 1; m < 16; m <<= 1) rs += __shfl_xor(rs, m, 32);
      lrun[r] = lrun[r] * al + rs;
      mrun[r] = mnew;
      alpha[r] = al;
      Pt[wave][r + 8 * hi][mloc]      = (_Float16)p0;
      Pt[wave][r + 8 * hi][16 + mloc] = (_Float16)p1;
    }
    // Pt is private to this wave; LDS ops from one wave are in-order, so no
    // barrier is needed between the P stores above and the A-frag load here.
    v16h pf = load_frag_a(&Pt[wave][mloc][hi * 16], &Pt[wave][mloc][hi * 16 + 8]);
#pragma unroll
    for (int dt = 0; dt < 4; ++dt) {
      int n = dt * 16 + mloc;
      v16h vf = load_frag_b(&Vt[n][hi * 16]);
#pragma unroll
      for (int r = 0; r < 8; ++r) oacc[dt][r] *= alpha[r];
      oacc[dt] = wmma_f16(pf, vf, oacc[dt]);
    }
  }

  // --- normalize and write [B,T,H,DV] f16 ---
#pragma unroll
  for (int dt = 0; dt < 4; ++dt) {
#pragma unroll
    for (int r = 0; r < 8; ++r) {
      int t  = tq0 + r + 8 * hi;
      int dv = dt * 16 + mloc;
      float val = oacc[dt][r] / lrun[r];
      o[((size_t)((b * Tc + t) * Hc + h)) * DVc + dv] = (_Float16)val;
    }
  }
}

// ---------------------------------------------------------------------------

extern "C" void kernel_launch(void* const* d_in, const int* in_sizes, int n_in,
                              void* d_out, int out_size, void* d_ws,
                              size_t ws_size, hipStream_t stream) {
  (void)in_sizes; (void)n_in; (void)out_size; (void)ws_size;
  const float* x      = (const float*)d_in[0];
  const float* wq     = (const float*)d_in[1];
  const float* wk     = (const float*)d_in[2];
  const float* wv     = (const float*)d_in[3];
  const float* w_proj = (const float*)d_in[4];
  const float* b_proj = (const float*)d_in[5];
  const float* w1     = (const float*)d_in[6];
  const float* b1     = (const float*)d_in[7];
  const float* w2     = (const float*)d_in[8];
  const float* b2     = (const float*)d_in[9];
  const float* g1     = (const float*)d_in[10];
  const float* be1    = (const float*)d_in[11];
  const float* g2     = (const float*)d_in[12];
  const float* be2    = (const float*)d_in[13];
  float* out = (float*)d_out;

  // workspace carve-up
  char* ws = (char*)d_ws;
  size_t off = 0;
  auto carve = [&](size_t bytes) -> char* {
    char* p = ws + off;
    off += (bytes + 255) & ~(size_t)255;
    return p;
  };
  _Float16* wqkv16 = (_Float16*)carve((size_t)3 * Ec * Ec * 2);
  _Float16* wp16   = (_Float16*)carve((size_t)Ec * Ec * 2);
  _Float16* w1_16  = (_Float16*)carve((size_t)Ec * E4c * 2);
  _Float16* w2_16  = (_Float16*)carve((size_t)E4c * Ec * 2);
  _Float16* h16    = (_Float16*)carve((size_t)BTc * Ec * 2);
  _Float16* q16    = (_Float16*)carve((size_t)BTc * Ec * 2);
  _Float16* k16    = (_Float16*)carve((size_t)BTc * Ec * 2);
  _Float16* v16    = (_Float16*)carve((size_t)BTc * Ec * 2);
  _Float16* attn16 = (_Float16*)carve((size_t)BTc * Ec * 2);
  float*    x2     = (float*)   carve((size_t)BTc * Ec * 4);
  _Float16* h2_16  = (_Float16*)carve((size_t)BTc * Ec * 2);
  _Float16* ff1_16 = (_Float16*)carve((size_t)BTc * E4c * 2);

  // 1) weight repack / downconvert
  repack_qkv_kernel<<<(3 * Ec * Ec) / 256, 256, 0, stream>>>(wq, wk, wv, wqkv16);
  cvt_f32_f16_kernel<<<(Ec * Ec) / 256, 256, 0, stream>>>(w_proj, wp16, Ec * Ec);
  cvt_f32_f16_kernel<<<(Ec * E4c) / 256, 256, 0, stream>>>(w1, w1_16, Ec * E4c);
  cvt_f32_f16_kernel<<<(E4c * Ec) / 256, 256, 0, stream>>>(w2, w2_16, E4c * Ec);

  // 2) LN1
  ln_kernel<<<BTc, 256, 0, stream>>>(x, g1, be1, h16);

  // 3) QKV projections (f16 out, [B,T,H,DV] via head-concat columns)
  dim3 blk(128);
  dim3 gqkv(Ec / BN, BTc / BM);
  gemm_kernel<0><<<gqkv, blk, 0, stream>>>(h16, wqkv16 + (size_t)0 * Ec * Ec,
                                           q16, nullptr, nullptr, BTc, Ec, Ec);
  gemm_kernel<0><<<gqkv, blk, 0, stream>>>(h16, wqkv16 + (size_t)1 * Ec * Ec,
                                           k16, nullptr, nullptr, BTc, Ec, Ec);
  gemm_kernel<0><<<gqkv, blk, 0, stream>>>(h16, wqkv16 + (size_t)2 * Ec * Ec,
                                           v16, nullptr, nullptr, BTc, Ec, Ec);

  // 4) causal flash attention (4 waves x 16 query rows per block)
  attn_kernel<<<Bc * Hc * (Tc / 64), 128, 0, stream>>>(q16, k16, v16, attn16);

  // 5) output projection + bias + residual -> x2 (f32)
  gemm_kernel<2><<<dim3(Ec / BN, BTc / BM), blk, 0, stream>>>(
      attn16, wp16, x2, b_proj, x, BTc, Ec, Ec);

  // 6) LN2
  ln_kernel<<<BTc, 256, 0, stream>>>(x2, g2, be2, h2_16);

  // 7) FFN1: bias + ReLU, f16 out
  gemm_kernel<1><<<dim3(E4c / BN, BTc / BM), blk, 0, stream>>>(
      h2_16, w1_16, ff1_16, b1, nullptr, BTc, E4c, Ec);

  // 8) FFN2: bias + residual(x2) -> final f32 output
  gemm_kernel<2><<<dim3(Ec / BN, BTc / BM), blk, 0, stream>>>(
      ff1_16, w2_16, out, b2, x2, BTc, Ec, E4c);
}
